// LSTMModel_2851858285091
// MI455X (gfx1250) — matrix operand order repaired and checked by reading, not compiled, not run on hardware
//
#include <hip/hip_runtime.h>
#include <cstdint>
#include <cstddef>

// ---------------------------------------------------------------------------
// Types for gfx1250 WMMA (wave32): D(16x16 f32) = A(16x32 bf16) x B(32x16 bf16) + C
// ---------------------------------------------------------------------------
typedef __attribute__((ext_vector_type(16))) __bf16 v16bf;
typedef __attribute__((ext_vector_type(8)))  __bf16 v8bf;
typedef __attribute__((ext_vector_type(4)))  __bf16 v4bf;
typedef __attribute__((ext_vector_type(8)))  float  v8f;

#define BSZ   128   // batch
#define TSZ   512   // timesteps
#define HID   256   // hidden
#define GATES 1024  // 4*HID
#define OUTD  128

union ABfrag { v16bf v; v8bf h[2]; };

// A-matrix fragment (16x32 bf16, M x K) from row-major [*, ld] at (m0, k0).
// Layout: lanes 0-15 -> row m0+lane, K k0..k0+15;
//         lanes 16-31 -> row m0+lane-16, K k0+16..k0+31 (matches load_b_frag).
__device__ inline v16bf load_a_frag(const __bf16* __restrict__ src, int ld,
                                    int m0, int k0, int lane) {
    const int m  = m0 + (lane & 15);
    const int kb = k0 + ((lane >> 4) << 4);          // +16 for upper half-wave
    const __bf16* p = src + (size_t)m * ld + kb;
    ABfrag f;
    f.h[0] = *(const v8bf*)(p);                      // 16B contiguous
    f.h[1] = *(const v8bf*)(p + 8);                  // 16B contiguous
    return f.v;
}

// B-matrix fragment (32x16 bf16, K x N) where B[k][n] = W[n0+n][k0+k] and W is
// row-major [N_total, ld] (i.e. we multiply by W^T, PyTorch weight convention).
// Layout: lanes 0-15 -> col n0+lane, K k0..k0+15 (contiguous in W's row);
//             lanes 16-31 -> col n0+lane-16, K k0+16..k0+31.
__device__ inline v16bf load_b_frag(const __bf16* __restrict__ W, int ld,
                                    int n0, int k0, int lane) {
    const int n  = n0 + (lane & 15);
    const int kk = k0 + ((lane >> 4) << 4);
    return *(const v16bf*)(W + (size_t)n * ld + kk); // 32B contiguous
}

__device__ inline float fast_sigmoid(float x) {
    return 1.0f / (1.0f + __expf(-x));
}
__device__ inline float fast_tanh(float x) {
    float e = __expf(-2.0f * x);
    return (1.0f - e) / (1.0f + e);
}

// ---------------------------------------------------------------------------
// x [B,T,128] f32  ->  xT [T,B,128] bf16   (time-major, contiguous per step)
// ---------------------------------------------------------------------------
__global__ __launch_bounds__(256) void transpose_x_kernel(
    const float* __restrict__ x, __bf16* __restrict__ xT) {
    const int idx = blockIdx.x * 256 + threadIdx.x;  // quad id, 2,097,152 total
    const int k4 = idx & 31;                         // 32 quads per 128-row
    const int bt = idx >> 5;                         // row id = b*T + t
    const int b  = bt >> 9;                          // / 512
    const int t  = bt & 511;
    const float4 v = ((const float4*)x)[idx];
    v4bf o; o.x = (__bf16)v.x; o.y = (__bf16)v.y; o.z = (__bf16)v.z; o.w = (__bf16)v.w;
    *(v4bf*)(xT + ((size_t)(t * BSZ + b) * 128) + k4 * 4) = o;
}

// fc_w [128,256] f32 -> bf16 (same layout)
__global__ __launch_bounds__(256) void convert_fcw_kernel(
    const float* __restrict__ w, __bf16* __restrict__ wb) {
    const int idx = blockIdx.x * 256 + threadIdx.x;  // 8192 quads
    const float4 v = ((const float4*)w)[idx];
    v4bf o; o.x = (__bf16)v.x; o.y = (__bf16)v.y; o.z = (__bf16)v.z; o.w = (__bf16)v.w;
    *(v4bf*)(wb + (size_t)idx * 4) = o;
}

// ---------------------------------------------------------------------------
// Persistent LSTM layer: 16 blocks x 256 threads (8 waves).
// Block bx owns hidden columns [16*bx, 16*bx+16); wave w owns batch rows
// [16*w, 16*w+16). Cell state lives in registers for the whole layer.
// ---------------------------------------------------------------------------
template <int IN_DIM>
__global__ __launch_bounds__(256) void lstm_layer_kernel(
    const __bf16* __restrict__ xT,      // [T, B, IN_DIM] bf16 (layer input)
    const float*  __restrict__ w_ih,    // [4H, IN_DIM] f32
    const float*  __restrict__ w_hh,    // [4H, H] f32
    const float*  __restrict__ b_ih,    // [4H] f32
    const float*  __restrict__ b_hh,    // [4H] f32
    __bf16*       __restrict__ ysT,     // [T, B, H] bf16 (layer output = h per step)
    unsigned int* __restrict__ cnt,     // [T] zeroed step counters
    float*        __restrict__ h_n,     // [B, H] f32 final h
    float*        __restrict__ c_n) {   // [B, H] f32 final c
    __shared__ __bf16 lds_wih[64 * IN_DIM];  // rows: gate g*16 + r  <-> W row g*H + j0 + r
    __shared__ __bf16 lds_whh[64 * HID];

    const int tid  = threadIdx.x;
    const int lane = tid & 31;
    const int wave = tid >> 5;
    const int j0   = blockIdx.x * 16;        // this block's hidden slice

    // Stage this block's weight slices into LDS (f32 -> bf16, once per layer).
    for (int e = tid; e < 64 * IN_DIM; e += 256) {
        const int lr = e / IN_DIM, k = e % IN_DIM;
        const int g = lr >> 4, r = lr & 15;
        lds_wih[lr * IN_DIM + k] = (__bf16)w_ih[(size_t)(g * HID + j0 + r) * IN_DIM + k];
    }
    for (int e = tid; e < 64 * HID; e += 256) {
        const int lr = e / HID, k = e % HID;
        const int g = lr >> 4, r = lr & 15;
        lds_whh[lr * HID + k] = (__bf16)w_hh[(size_t)(g * HID + j0 + r) * HID + k];
    }
    __syncthreads();

    const int col = lane & 15;               // this lane's column within the tile
    float bias[4];
#pragma unroll
    for (int g = 0; g < 4; ++g)
        bias[g] = b_ih[g * HID + j0 + col] + b_hh[g * HID + j0 + col];

    const int m0 = wave * 16;                // batch-row tile
    v8f c_st = {};                           // persistent cell state (registers)

    for (int t = 0; t < TSZ; ++t) {
        v8f acc[4] = {};                     // i, f, g, o accumulators (f32)

        // ---- input projection: x_t @ W_ih^T ----
        const __bf16* xrow = xT + (size_t)t * BSZ * IN_DIM;
        if (t + 1 < TSZ)                     // pull next step into cache early
            __builtin_prefetch(xT + (size_t)(t + 1) * BSZ * IN_DIM +
                               (size_t)(m0 + col) * IN_DIM, 0, 1);
#pragma unroll
        for (int k0 = 0; k0 < IN_DIM; k0 += 32) {
            const v16bf a = load_a_frag(xrow, IN_DIM, m0, k0, lane);
#pragma unroll
            for (int g = 0; g < 4; ++g) {
                const v16bf b = load_b_frag(lds_wih, IN_DIM, g * 16, k0, lane);
                acc[g] = __builtin_amdgcn_wmma_f32_16x16x32_bf16(
                    false, a, false, b, (short)0, acc[g], false, false);
            }
        }
        // ---- recurrent projection: h_{t-1} @ W_hh^T (h0 == 0 -> skip) ----
        if (t > 0) {
            const __bf16* hrow = ysT + (size_t)(t - 1) * BSZ * HID;
#pragma unroll
            for (int k0 = 0; k0 < HID; k0 += 32) {
                const v16bf a = load_a_frag(hrow, HID, m0, k0, lane);
#pragma unroll
                for (int g = 0; g < 4; ++g) {
                    const v16bf b = load_b_frag(lds_whh, HID, g * 16, k0, lane);
                    acc[g] = __builtin_amdgcn_wmma_f32_16x16x32_bf16(
                        false, a, false, b, (short)0, acc[g], false, false);
                }
            }
        }

        // ---- gates, state update, emit h (D-layout: lane<16 rows m0+j, else m0+8+j)
        __bf16* hout = ysT + (size_t)t * BSZ * HID;
        const int rbase = m0 + ((lane >> 4) << 3);
#pragma unroll
        for (int j = 0; j < 8; ++j) {
            const float iv = fast_sigmoid(acc[0][j] + bias[0]);
            const float fv = fast_sigmoid(acc[1][j] + bias[1]);
            const float gv = fast_tanh   (acc[2][j] + bias[2]);
            const float ov = fast_sigmoid(acc[3][j] + bias[3]);
            const float c  = fv * c_st[j] + iv * gv;
            c_st[j] = c;
            const float h = ov * fast_tanh(c);
            hout[(size_t)(rbase + j) * HID + j0 + col] = (__bf16)h;
            if (t == TSZ - 1) {
                h_n[(size_t)(rbase + j) * HID + j0 + col] = h;
                c_n[(size_t)(rbase + j) * HID + j0 + col] = c;
            }
        }

        // ---- device-wide step barrier across the 16 resident blocks ----
        __syncthreads();
        __threadfence();                                 // release h stores
        if (tid == 0) {
            __hip_atomic_fetch_add(&cnt[t], 1u, __ATOMIC_RELEASE,
                                   __HIP_MEMORY_SCOPE_AGENT);
            while (__hip_atomic_load(&cnt[t], __ATOMIC_ACQUIRE,
                                     __HIP_MEMORY_SCOPE_AGENT) < 16u)
                __builtin_amdgcn_s_sleep(2);
        }
        __syncthreads();
        __threadfence();                                 // acquire others' h
    }
}

// ---------------------------------------------------------------------------
// FC: logits[b*T+t, n] = ysT[t*B+b, :] @ fc_w^T + fc_b   (M=65536, N=128, K=256)
// 8 waves/block, one 16x16 output tile per wave.
// ---------------------------------------------------------------------------
__global__ __launch_bounds__(256) void fc_kernel(
    const __bf16* __restrict__ ysT, const __bf16* __restrict__ fcw,
    const float* __restrict__ fc_b, float* __restrict__ out) {
    const int lane = threadIdx.x & 31;
    const int wave = threadIdx.x >> 5;
    const int tile = blockIdx.x * 8 + wave;
    const int n0 = (tile & 7) * 16;          // 8 N tiles
    const int m0 = (tile >> 3) * 16;         // 4096 M tiles
    v8f acc = {};
#pragma unroll
    for (int k0 = 0; k0 < HID; k0 += 32) {
        const v16bf a = load_a_frag(ysT, HID, m0, k0, lane);
        const v16bf b = load_b_frag(fcw, HID, n0, k0, lane);
        acc = __builtin_amdgcn_wmma_f32_16x16x32_bf16(
            false, a, false, b, (short)0, acc, false, false);
    }
    const int coln = n0 + (lane & 15);
    const float bias = fc_b[coln];
    const int rbase = m0 + ((lane >> 4) << 3);
#pragma unroll
    for (int j = 0; j < 8; ++j) {
        const int r = rbase + j;             // r = t*B + b (time-major)
        const int t = r >> 7;                // / 128
        const int b = r & 127;
        out[((size_t)b * TSZ + t) * OUTD + coln] = acc[j] + bias;
    }
}

// ---------------------------------------------------------------------------
extern "C" void kernel_launch(void* const* d_in, const int* in_sizes, int n_in,
                              void* d_out, int out_size, void* d_ws, size_t ws_size,
                              hipStream_t stream) {
    const float* x     = (const float*)d_in[0];
    const float* w_ih[3] = { (const float*)d_in[1], (const float*)d_in[5], (const float*)d_in[9]  };
    const float* w_hh[3] = { (const float*)d_in[2], (const float*)d_in[6], (const float*)d_in[10] };
    const float* b_ih[3] = { (const float*)d_in[3], (const float*)d_in[7], (const float*)d_in[11] };
    const float* b_hh[3] = { (const float*)d_in[4], (const float*)d_in[8], (const float*)d_in[12] };
    const float* fc_w = (const float*)d_in[13];
    const float* fc_b = (const float*)d_in[14];

    // Workspace layout (bytes):
    char* ws = (char*)d_ws;
    unsigned int* cnt = (unsigned int*)ws;                       // 3*512 counters
    __bf16* xT0  = (__bf16*)(ws + 8192);                         // 16 MB  [T,B,128]
    __bf16* ysA  = (__bf16*)(ws + 8192 + 16777216);              // 32 MB  [T,B,256]
    __bf16* ysB  = (__bf16*)(ws + 8192 + 16777216 + 33554432);   // 32 MB  [T,B,256]
    __bf16* fcwb = (__bf16*)(ws + 8192 + 16777216 + 2u*33554432);// 64 KB

    float* out = (float*)d_out;                  // logits [B*T, 128]
    float* h_n = out + (size_t)BSZ * TSZ * OUTD; // [3, B, H]
    float* c_n = h_n + (size_t)3 * BSZ * HID;    // [3, B, H]

    hipMemsetAsync(cnt, 0, 3 * TSZ * sizeof(unsigned int), stream);
    transpose_x_kernel<<<8192, 256, 0, stream>>>(x, xT0);
    convert_fcw_kernel<<<32, 256, 0, stream>>>(fc_w, fcwb);

    lstm_layer_kernel<128><<<16, 256, 0, stream>>>(
        xT0, w_ih[0], w_hh[0], b_ih[0], b_hh[0], ysA, cnt + 0 * TSZ,
        h_n + 0 * BSZ * HID, c_n + 0 * BSZ * HID);
    lstm_layer_kernel<256><<<16, 256, 0, stream>>>(
        ysA, w_ih[1], w_hh[1], b_ih[1], b_hh[1], ysB, cnt + 1 * TSZ,
        h_n + 1 * BSZ * HID, c_n + 1 * BSZ * HID);
    lstm_layer_kernel<256><<<16, 256, 0, stream>>>(
        ysB, w_ih[2], w_hh[2], b_ih[2], b_hh[2], ysA, cnt + 2 * TSZ,
        h_n + 2 * BSZ * HID, c_n + 2 * BSZ * HID);

    fc_kernel<<<4096, 256, 0, stream>>>(ysA, fcwb, fc_b, out);
}
